// bwd_mpgnn_64793876627815
// MI455X (gfx1250) — matrix operand order repaired and checked by reading, not compiled, not run on hardware
//
#include <hip/hip_runtime.h>

// GNN message-passing stack for MI455X (gfx1250, wave32, WMMA).
// Dense math: v_wmma_f32_16x16x32_f16, f32 accumulate.
// Per-wave tile 64x64 (16 WMMA : 16 ds_load_b128 per K-step).
// Staging: global_load_async_to_lds_b128 with LDS double-buffering (ASYNCcnt).

typedef __attribute__((ext_vector_type(16))) _Float16 v16h;
typedef __attribute__((ext_vector_type(8)))  _Float16 v8h;
typedef __attribute__((ext_vector_type(8)))  float    v8f;
typedef __attribute__((ext_vector_type(4)))  int      v4i;

#define LVLS 6
#define NNODES 32768
#define FANIN 8
#define FDIM 32
#define HDIM 256

#define BLK_M 512   // 8 waves x 64 rows
#define BLK_N 64
#define BLK_K 32
#define MI 4        // 16-row m-frags per wave
#define NJ 4        // 16-col n-frags per wave
#define LDSS 40     // padded LDS row stride (halfs): 80B, 16B-aligned, bank-friendly

#if __has_builtin(__builtin_amdgcn_global_load_async_to_lds_b128)
#define HAVE_ASYNC_LDS 1
#else
#define HAVE_ASYNC_LDS 0
#endif

#if HAVE_ASYNC_LDS
typedef __attribute__((address_space(1))) v4i gv4i;   // clang prints AS1 as __device__
typedef __attribute__((address_space(3))) v4i lv4i;   // LDS
__device__ __forceinline__ void async_ld_b128(const void* g, void* l) {
    __builtin_amdgcn_global_load_async_to_lds_b128((gv4i*)g, (lv4i*)l, 0, 0);
}
__device__ __forceinline__ void wait_async0() {
#if __has_builtin(__builtin_amdgcn_s_wait_asynccnt)
    __builtin_amdgcn_s_wait_asynccnt(0);
#else
    asm volatile("s_wait_asynccnt 0x0" ::: "memory");
#endif
}
#endif

__device__ __forceinline__ float dev_tanh(float x) {
#if __has_builtin(__builtin_amdgcn_tanhf)
    return __builtin_amdgcn_tanhf(x);     // v_tanh_f32 (CDNA5 trans op)
#else
    x = fminf(fmaxf(x, -15.f), 15.f);
    float e = __expf(2.f * x);            // v_exp_f32
    return (e - 1.f) / (e + 1.f);
#endif
}

// C = tanh(A @ W^T + b) [+ res]; A:[M][lda] f16, W:[Nout][ldw] f16 (row n = output n),
// out16:[M][ldo] f16 always; out32 optional f32. Residual added AFTER tanh (h2 + x fold).
__global__ __launch_bounds__(256) void gemm_tanh_kernel(
    const _Float16* __restrict__ A, int lda,
    const _Float16* __restrict__ W, int ldw,
    const float* __restrict__ bias,
    const _Float16* __restrict__ res, int ldres, int rescols,
    _Float16* __restrict__ out16, int ldo,
    float* __restrict__ out32, int ldo32,
    int K)
{
    __shared__ _Float16 As[2][BLK_M][LDSS];
    __shared__ _Float16 Ws[2][BLK_N][LDSS];

    const int tid    = threadIdx.x;
    const int lane   = tid & 31;
    const int wave   = tid >> 5;
    const int l16    = lane & 15;
    const int hi     = lane >> 4;          // lane half-select
    const int baseA  = hi * 8;             // A frag: elems0-7 = K base..+7, elems8-15 = K base+16..+23
    const int baseB  = hi * 8;             // B frag: elems0-7 = K base..+7, elems8-15 = K base+16..+23
    const int blockM = blockIdx.x * BLK_M;
    const int blockN = blockIdx.y * BLK_N;
    const int wrow   = tid >> 2;           // 0..63 (W stage row)
    const int wseg   = (tid & 3) * 8;      // half offset

    v8f acc[MI][NJ];
    #pragma unroll
    for (int i = 0; i < MI; i++)
        #pragma unroll
        for (int j = 0; j < NJ; j++)
            #pragma unroll
            for (int r = 0; r < 8; r++)
                acc[i][j][r] = 0.0f;

#if HAVE_ASYNC_LDS
    // issue async DMA for one K-tile into LDS buffer `buf` (no VGPR round-trip)
    auto stage = [&](int buf, int k0) {
        #pragma unroll
        for (int s = 0; s < 2; s++) {
            int row = tid + s * 256;
            const _Float16* g = A + (size_t)(blockM + row) * lda + k0;
            #pragma unroll
            for (int jj = 0; jj < 4; jj++)
                async_ld_b128(g + jj * 8, &As[buf][row][jj * 8]);
        }
        async_ld_b128(W + (size_t)(blockN + wrow) * ldw + k0 + wseg,
                      &Ws[buf][wrow][wseg]);
    };
#else
    // fallback: bulk loads into distinct regs, then bulk LDS stores
    auto stage = [&](int buf, int k0) {
        uint4 t[9];
        #pragma unroll
        for (int s = 0; s < 2; s++) {
            const uint4* src = (const uint4*)(A + (size_t)(blockM + tid + s * 256) * lda + k0);
            #pragma unroll
            for (int jj = 0; jj < 4; jj++) t[s * 4 + jj] = src[jj];
        }
        t[8] = *(const uint4*)(W + (size_t)(blockN + wrow) * ldw + k0 + wseg);
        #pragma unroll
        for (int s = 0; s < 2; s++) {
            uint4* dst = (uint4*)&As[buf][tid + s * 256][0];
            #pragma unroll
            for (int jj = 0; jj < 4; jj++) dst[jj] = t[s * 4 + jj];
        }
        *(uint4*)&Ws[buf][wrow][wseg] = t[8];
    };
#endif

    stage(0, 0);
    int buf = 0;

    for (int k0 = 0; k0 < K; k0 += BLK_K) {
#if HAVE_ASYNC_LDS
        wait_async0();          // my wave's DMA into As[buf]/Ws[buf] done
#endif
        __syncthreads();        // all waves' staging of buf visible; prev reads of buf^1 retired
        if (k0 + BLK_K < K)
            stage(buf ^ 1, k0 + BLK_K);   // next tile DMA overlaps this tile's WMMAs

        v16h afrag[MI], bfrag[NJ];
        #pragma unroll
        for (int i = 0; i < MI; i++) {
            int r = wave * 64 + i * 16 + l16;
            v8h lo = *(const v8h*)&As[buf][r][baseA];
            v8h hh = *(const v8h*)&As[buf][r][baseA + 16];
            afrag[i] = __builtin_shufflevector(lo, hh, 0,1,2,3,4,5,6,7,8,9,10,11,12,13,14,15);
        }
        #pragma unroll
        for (int j = 0; j < NJ; j++) {
            int r = j * 16 + l16;
            v8h lo = *(const v8h*)&Ws[buf][r][baseB];
            v8h hh = *(const v8h*)&Ws[buf][r][baseB + 16];
            bfrag[j] = __builtin_shufflevector(lo, hh, 0,1,2,3,4,5,6,7,8,9,10,11,12,13,14,15);
        }

        #pragma unroll
        for (int i = 0; i < MI; i++)
            #pragma unroll
            for (int j = 0; j < NJ; j++)
                acc[i][j] = __builtin_amdgcn_wmma_f32_16x16x32_f16(
                    false, afrag[i], false, bfrag[j],
                    (short)0, acc[i][j], false, false);

        buf ^= 1;
    }

    // --- epilogue: bias + tanh (+ residual), f16 store, optional f32 store ---
    #pragma unroll
    for (int j = 0; j < NJ; j++) {
        int n = blockN + j * 16 + l16;
        float bn = bias[n];
        #pragma unroll
        for (int i = 0; i < MI; i++) {
            int mbase = blockM + wave * 64 + i * 16 + hi * 8;
            #pragma unroll
            for (int r = 0; r < 8; r++) {
                int m = mbase + r;
                float y = dev_tanh(acc[i][j][r] + bn);
                if (res != nullptr && n < rescols)
                    y += (float)res[(size_t)m * ldres + n];
                out16[(size_t)m * ldo + n] = (_Float16)y;
                if (out32 != nullptr)
                    out32[(size_t)m * ldo32 + n] = y;
            }
        }
    }
}

// msgs[m] = sum_j emb[idx[m][j]]  (K=8 gather-sum; emb table is 16MB f16 -> L2-resident)
__global__ __launch_bounds__(256) void gather_sum_kernel(
    const _Float16* __restrict__ emb,   // [N][H]
    const int* __restrict__ idx,        // [N][8]
    _Float16* __restrict__ msgs)        // [N][H]
{
    int gid = blockIdx.x * blockDim.x + threadIdx.x;   // N * H/8 threads
    int m = gid >> 5;                                  // H/8 = 32 chunks/row
    int c = (gid & 31) << 3;                           // half offset
    const int* ip = idx + (size_t)m * FANIN;
    float acc[8];
    #pragma unroll
    for (int t = 0; t < 8; t++) acc[t] = 0.f;
    #pragma unroll
    for (int j = 0; j < FANIN; j++) {
        v8h v = *(const v8h*)(emb + (size_t)ip[j] * HDIM + c);
        #pragma unroll
        for (int t = 0; t < 8; t++) acc[t] += (float)v[t];
    }
    v8h o;
    #pragma unroll
    for (int t = 0; t < 8; t++) o[t] = (_Float16)acc[t];
    *(v8h*)(msgs + (size_t)m * HDIM + c) = o;
}

__global__ __launch_bounds__(256) void cvt_f32_f16_kernel(
    const float* __restrict__ s, _Float16* __restrict__ d, int n)
{
    int i = blockIdx.x * blockDim.x + threadIdx.x;
    if (i < n) d[i] = (_Float16)s[i];
}

extern "C" void kernel_launch(void* const* d_in, const int* in_sizes, int n_in,
                              void* d_out, int out_size, void* d_ws, size_t ws_size,
                              hipStream_t stream)
{
    const float* feats = (const float*)d_in[0];
    const int*   pred  = (const int*)d_in[1];
    const float* We    = (const float*)d_in[2];
    const float* be    = (const float*)d_in[3];
    const float* rsW1  = (const float*)d_in[4];
    const float* rsb1  = (const float*)d_in[5];
    const float* rsW2  = (const float*)d_in[6];
    const float* rsb2  = (const float*)d_in[7];
    const float* rsW3  = (const float*)d_in[8];
    const float* rsb3  = (const float*)d_in[9];
    const float* rcW1  = (const float*)d_in[10];
    const float* rcb1  = (const float*)d_in[11];
    const float* rcW2  = (const float*)d_in[12];
    const float* rcb2  = (const float*)d_in[13];
    const float* rcW3  = (const float*)d_in[14];
    const float* rcb3  = (const float*)d_in[15];
    float* out = (float*)d_out;

    const int N = NNODES, H = HDIM, F = FDIM;

    // --- workspace bump allocator (256B aligned) ---
    char* p = (char*)d_ws;
    auto alloc = [&](size_t bytes) -> char* {
        char* r = p; p += (bytes + 255) & ~(size_t)255; return r;
    };
    _Float16* feats16 = (_Float16*)alloc((size_t)LVLS * N * F * 2);
    _Float16* We16    = (_Float16*)alloc((size_t)H * F * 2);
    _Float16* rsW1h   = (_Float16*)alloc((size_t)12 * H * H * 2);
    _Float16* rsW2h   = (_Float16*)alloc((size_t)12 * H * H * 2);
    _Float16* rsW3h   = (_Float16*)alloc((size_t)12 * H * H * 2);
    _Float16* rcW1h   = (_Float16*)alloc((size_t)6 * 512 * 512 * 2);
    _Float16* rcW2h   = (_Float16*)alloc((size_t)6 * 512 * 512 * 2);
    _Float16* rcW3h   = (_Float16*)alloc((size_t)6 * 256 * 512 * 2);
    _Float16* ebuf    = (_Float16*)alloc((size_t)N * H * 2);
    _Float16* sa      = (_Float16*)alloc((size_t)N * H * 2);
    _Float16* sb      = (_Float16*)alloc((size_t)N * H * 2);
    _Float16* msgs    = (_Float16*)alloc((size_t)N * H * 2);
    _Float16* h1b     = (_Float16*)alloc((size_t)N * 512 * 2);
    _Float16* h2b     = (_Float16*)alloc((size_t)N * 512 * 2);
    _Float16* x2b     = (_Float16*)alloc((size_t)N * 512 * 2);
    _Float16* embA    = (_Float16*)alloc((size_t)N * H * 2);
    _Float16* embB    = (_Float16*)alloc((size_t)N * H * 2);

    auto cvt = [&](const float* s, _Float16* d, int n) {
        cvt_f32_f16_kernel<<<(n + 255) / 256, 256, 0, stream>>>(s, d, n);
    };
    cvt(feats, feats16, LVLS * N * F);
    cvt(We,   We16,  H * F);
    cvt(rsW1, rsW1h, 12 * H * H);
    cvt(rsW2, rsW2h, 12 * H * H);
    cvt(rsW3, rsW3h, 12 * H * H);
    cvt(rcW1, rcW1h, 6 * 512 * 512);
    cvt(rcW2, rcW2h, 6 * 512 * 512);
    cvt(rcW3, rcW3h, 6 * 256 * 512);

    auto gemm = [&](const _Float16* A, int lda, const _Float16* Wp, int ldw,
                    const float* bias, const _Float16* resp, int ldres, int rescols,
                    _Float16* o16, int ldo, float* o32, int ldo32,
                    int Nout, int Kdim) {
        dim3 grid(N / BLK_M, Nout / BLK_N);
        gemm_tanh_kernel<<<grid, 256, 0, stream>>>(A, lda, Wp, ldw, bias,
                                                   resp, ldres, rescols,
                                                   o16, ldo, o32, ldo32, Kdim);
    };

    // level 0: embeds0 = tanh(feats0 @ We.T + be)
    gemm(feats16, F, We16, F, be, nullptr, 0, 0, embA, H, out, H, H, F);

    _Float16* ep = embA;   // previous-level embeds (f16, for gather)
    _Float16* ec = embB;

    const size_t so = (size_t)H * H;          // small weight stride
    const size_t bo1 = (size_t)512 * 512;     // big W1/W2 stride
    const size_t bo3 = (size_t)256 * 512;     // big W3 stride

    for (int l = 1; l < LVLS; l++) {
        int d   = (l - 1 < 2) ? (l - 1) : 2;
        int bC  = 3 + d;    // rc idx: comb branch
        int sC  = 9 + d;    // rs idx: comb1
        int bN_ = d;        // rc idx: node branch
        int sN_ = 6 + d;    // rs idx: node1
        int m1  = 2 * d, m2 = 2 * d + 1;      // rs idx: mp, mp1

        const _Float16* fl = feats16 + (size_t)l * N * F;

        // e = tanh(feats @ We.T + be)
        gemm(fl, F, We16, F, be, nullptr, 0, 0, ebuf, H, nullptr, 0, H, F);

        // big(3+d) on x=[e|0]: layer1 uses only left half of W1 (K=256)
        gemm(ebuf, H, rcW1h + bC * bo1, 512, rcb1 + bC * 512, nullptr, 0, 0, h1b, 512, nullptr, 0, 512, 256);
        gemm(h1b, 512, rcW2h + bC * bo1, 512, rcb2 + bC * 512, ebuf, H, H, h2b, 512, nullptr, 0, 512, 512);
        gemm(h2b, 512, rcW3h + bC * bo3, 512, rcb3 + bC * 256, nullptr, 0, 0, sa, H, nullptr, 0, H, 512);
        // small(9+d): e2 -> left half of x2
        gemm(sa, H, rsW1h + sC * so, H, rsb1 + sC * H, nullptr, 0, 0, sb, H, nullptr, 0, H, H);
        gemm(sb, H, rsW2h + sC * so, H, rsb2 + sC * H, sa, H, H, sa, H, nullptr, 0, H, H);
        gemm(sa, H, rsW3h + sC * so, H, rsb3 + sC * H, nullptr, 0, 0, x2b, 512, nullptr, 0, H, H);

        // msgs = sum_k embeds[l-1][pred_idx]
        gather_sum_kernel<<<(N * (H / 8)) / 256, 256, 0, stream>>>(
            ep, pred + (size_t)(l - 1) * N * FANIN, msgs);

        // r = tanh(small(2d, msgs))
        gemm(msgs, H, rsW1h + m1 * so, H, rsb1 + m1 * H, nullptr, 0, 0, sa, H, nullptr, 0, H, H);
        gemm(sa, H, rsW2h + m1 * so, H, rsb2 + m1 * H, msgs, H, H, msgs, H, nullptr, 0, H, H);
        gemm(msgs, H, rsW3h + m1 * so, H, rsb3 + m1 * H, nullptr, 0, 0, sb, H, nullptr, 0, H, H);
        // r = tanh(small(2d+1, r)) -> right half of x2
        gemm(sb, H, rsW1h + m2 * so, H, rsb1 + m2 * H, nullptr, 0, 0, sa, H, nullptr, 0, H, H);
        gemm(sa, H, rsW2h + m2 * so, H, rsb2 + m2 * H, sb, H, H, sb, H, nullptr, 0, H, H);
        gemm(sb, H, rsW3h + m2 * so, H, rsb3 + m2 * H, nullptr, 0, 0, x2b + 256, 512, nullptr, 0, H, H);

        // big(d) on x2
        gemm(x2b, 512, rcW1h + bN_ * bo1, 512, rcb1 + bN_ * 512, nullptr, 0, 0, h1b, 512, nullptr, 0, 512, 512);
        gemm(h1b, 512, rcW2h + bN_ * bo1, 512, rcb2 + bN_ * 512, x2b, 512, 512, h2b, 512, nullptr, 0, 512, 512);
        gemm(h2b, 512, rcW3h + bN_ * bo3, 512, rcb3 + bN_ * 256, nullptr, 0, 0, sa, H, nullptr, 0, H, 512);
        // small(6+d): final embeds -> f16 (next gather) + f32 (d_out)
        gemm(sa, H, rsW1h + sN_ * so, H, rsb1 + sN_ * H, nullptr, 0, 0, sb, H, nullptr, 0, H, H);
        gemm(sb, H, rsW2h + sN_ * so, H, rsb2 + sN_ * H, sa, H, H, sa, H, nullptr, 0, H, H);
        gemm(sa, H, rsW3h + sN_ * so, H, rsb3 + sN_ * H, nullptr, 0, 0, ec, H,
             out + (size_t)l * N * H, H, H, H);

        _Float16* t = ep; ep = ec; ec = t;
    }
    (void)in_sizes; (void)n_in; (void)out_size; (void)ws_size;
}
